// GCN_ST_42700564857468
// MI455X (gfx1250) — compile-verified, matrix operand order checked by reading, not run on hardware
//
#include <hip/hip_runtime.h>
#include <hip/hip_bf16.h>

typedef __attribute__((ext_vector_type(16))) _Float16 v16h;
typedef __attribute__((ext_vector_type(8)))  float    v8f;

#define NN   512     // nodes
#define NE   4096    // edges
#define BB   32      // batch
#define HH   32      // hidden channels
#define SKC  256     // skip channels
#define ECC  512     // end channels
#define OUTF 12      // output features
#define AEMB 10      // adaptive embedding dim
#define NEGBIG (-1e30f)
#define NLAY 8

__device__ __forceinline__ v8f wmma_f16(v16h a, v16h b, v8f c) {
    return __builtin_amdgcn_wmma_f32_16x16x32_f16(false, a, false, b, (short)0, c, false, false);
}

// fast transcendentals on the hw v_exp/v_rcp path (no libm slow-path branches)
__device__ __forceinline__ float fast_sigmoid(float x) {
    return __builtin_amdgcn_rcpf(1.f + __expf(-x));
}
__device__ __forceinline__ float fast_tanh(float x) {
    float t = __expf(-2.f * fabsf(x));          // in (0,1], never overflows
    float r = (1.f - t) * __builtin_amdgcn_rcpf(1.f + t);
    return copysignf(r, x);
}

// Pack 8 consecutive f32 (16B-aligned) into slots [off..off+7] as f16.
__device__ __forceinline__ void pack8(v16h& d, int off, const float* __restrict__ p) {
    float4 x = ((const float4*)p)[0];
    float4 y = ((const float4*)p)[1];
    d[off + 0] = (_Float16)x.x; d[off + 1] = (_Float16)x.y;
    d[off + 2] = (_Float16)x.z; d[off + 3] = (_Float16)x.w;
    d[off + 4] = (_Float16)y.x; d[off + 5] = (_Float16)y.y;
    d[off + 6] = (_Float16)y.z; d[off + 7] = (_Float16)y.w;
}
__device__ __forceinline__ void pack8_scale(v16h& d, int off, const float* __restrict__ p, float s) {
    float4 x = ((const float4*)p)[0];
    float4 y = ((const float4*)p)[1];
    d[off + 0] = (_Float16)(x.x * s); d[off + 1] = (_Float16)(x.y * s);
    d[off + 2] = (_Float16)(x.z * s); d[off + 3] = (_Float16)(x.w * s);
    d[off + 4] = (_Float16)(y.x * s); d[off + 5] = (_Float16)(y.y * s);
    d[off + 6] = (_Float16)(y.z * s); d[off + 7] = (_Float16)(y.w * s);
}
__device__ __forceinline__ void pack8_relu(v16h& d, int off, const float* __restrict__ p) {
    float4 x = ((const float4*)p)[0];
    float4 y = ((const float4*)p)[1];
    d[off + 0] = (_Float16)fmaxf(x.x, 0.f); d[off + 1] = (_Float16)fmaxf(x.y, 0.f);
    d[off + 2] = (_Float16)fmaxf(x.z, 0.f); d[off + 3] = (_Float16)fmaxf(x.w, 0.f);
    d[off + 4] = (_Float16)fmaxf(y.x, 0.f); d[off + 5] = (_Float16)fmaxf(y.y, 0.f);
    d[off + 6] = (_Float16)fmaxf(y.z, 0.f); d[off + 7] = (_Float16)fmaxf(y.w, 0.f);
}

// ---------------------------------------------------------------- utilities
__global__ void k_zero(float* p, int n) {
    int i = blockIdx.x * blockDim.x + threadIdx.x;
    if (i < n) p[i] = 0.f;
}

// Repack weights so the WMMA B operand is contiguous in K:
//  fwT/gwT: [L,c,tap,ci] ; gcwT: [L,n,k]
__global__ void k_prep_w(const float* __restrict__ fw, const float* __restrict__ gw,
                         const float* __restrict__ gcw,
                         float* fwT, float* gwT, float* gcwT) {
    int idx = blockIdx.x * blockDim.x + threadIdx.x;
    const int totalFG = NLAY * HH * 2 * HH;  // 16384
    if (idx < totalFG) {
        int ci = idx & 31;
        int tap = (idx >> 5) & 1;
        int c = (idx >> 6) & 31;
        int l = idx >> 11;
        fwT[idx] = fw[((l * HH + c) * HH + ci) * 2 + tap];
        gwT[idx] = gw[((l * HH + c) * HH + ci) * 2 + tap];
    }
    if (idx < NLAY * HH * HH) {  // 8192
        int k = idx & 31;
        int n = (idx >> 5) & 31;
        int l = idx >> 10;
        gcwT[idx] = gcw[l * HH * HH + k * HH + n];
    }
}

// ------------------------------------------------- graph preprocessing
__global__ void k_scatter(const int* __restrict__ src, const int* __restrict__ dst,
                          unsigned char* adj, float* degout, float* degin) {
    int e = blockIdx.x * blockDim.x + threadIdx.x;
    if (e >= NE) return;
    int s = src[e], d = dst[e];
    adj[s * NN + d] = 1;
    atomicAdd(degout + s, 1.f);
    atomicAdd(degin + d, 1.f);
}

__global__ void k_rownorm(const unsigned char* __restrict__ adj,
                          const float* __restrict__ nv1, const float* __restrict__ nv2,
                          float* rowmax, float* rowsum) {
    __shared__ float sc[NN];
    __shared__ float red[256];
    int i = blockIdx.x;
    for (int j = threadIdx.x; j < NN; j += blockDim.x) {
        float v = NEGBIG;
        if (adj[i * NN + j]) {
            float acc = 0.f;
#pragma unroll
            for (int a = 0; a < AEMB; a++) acc += nv1[i * AEMB + a] * nv2[a * NN + j];
            v = fmaxf(acc, 0.f);
        }
        sc[j] = v;
    }
    __syncthreads();
    float m = NEGBIG;
    for (int j = threadIdx.x; j < NN; j += blockDim.x) m = fmaxf(m, sc[j]);
    red[threadIdx.x] = m;
    __syncthreads();
    for (int s = blockDim.x / 2; s > 0; s >>= 1) {
        if (threadIdx.x < s) red[threadIdx.x] = fmaxf(red[threadIdx.x], red[threadIdx.x + s]);
        __syncthreads();
    }
    m = red[0];
    __syncthreads();
    float sum = 0.f;
    for (int j = threadIdx.x; j < NN; j += blockDim.x) {
        float v = sc[j];
        if (v > 0.5f * NEGBIG) sum += __expf(v - m);  // v-m <= 0
    }
    red[threadIdx.x] = sum;
    __syncthreads();
    for (int s = blockDim.x / 2; s > 0; s >>= 1) {
        if (threadIdx.x < s) red[threadIdx.x] += red[threadIdx.x + s];
        __syncthreads();
    }
    if (threadIdx.x == 0) {
        rowmax[i] = m;
        rowsum[i] = (red[0] > 0.f) ? red[0] : 1.f;
    }
}

__global__ void k_ew(const int* __restrict__ src, const int* __restrict__ dst,
                     const float* __restrict__ nv1, const float* __restrict__ nv2,
                     const float* __restrict__ rowmax, const float* __restrict__ rowsum,
                     float* ew) {
    int e = blockIdx.x * blockDim.x + threadIdx.x;
    if (e >= NE) return;
    int s = src[e], d = dst[e];
    float acc = 0.f;
#pragma unroll
    for (int a = 0; a < AEMB; a++) acc += nv1[s * AEMB + a] * nv2[a * NN + d];
    float scv = fmaxf(acc, 0.f);
    ew[e] = 1.f + __expf(scv - rowmax[s]) * __builtin_amdgcn_rcpf(rowsum[s]);
}

__global__ void k_norms(const float* degout, const float* degin, float* nsv, float* ndv) {
    int i = blockIdx.x * blockDim.x + threadIdx.x;
    if (i >= NN) return;
    nsv[i] = rsqrtf(fmaxf(degout[i], 1.f));
    ndv[i] = rsqrtf(fmaxf(degin[i], 1.f));
}

// --------------------------------------- start conv: [B,N,13,HH], t=0 is pad
__global__ void k_start(const float* __restrict__ feat, const float* __restrict__ sw,
                        const float* __restrict__ sb, float* h) {
    int idx = blockIdx.x * blockDim.x + threadIdx.x;
    const int total = BB * NN * 13 * HH;
    if (idx >= total) return;
    int c = idx & 31;
    int t = (idx >> 5) % 13;
    int m = idx / (13 * HH);  // b*NN + n
    int b = m >> 9, n = m & 511;
    float x = (t == 0) ? 0.f : feat[(b * 12 + (t - 1)) * NN + n];
    h[idx] = sw[c] * x + sb[c];
}

// ---------------- gated activation unit as WMMA: K=64 ([h_t | h_{t+d}]), N=32
__global__ void k_gau_wmma(const float* __restrict__ h, const float* __restrict__ fwT,
                           const float* __restrict__ fb, const float* __restrict__ gwT,
                           const float* __restrict__ gb, float* gau, int T, int d, int l) {
    int Tp = T - d;
    int tile = blockIdx.x * 4 + (threadIdx.x >> 5);
    int tm = tile >> 1, tn = tile & 1;   // 2 N-tiles (32 channels)
    int lane = threadIdx.x & 31;
    int mrow = tm * 16 + (lane & 15);    // m = (b*NN+n)*Tp + t
    int t = mrow % Tp;
    int bn = mrow / Tp;
    const float* hrow0 = h + (bn * T + t) * HH;
    const float* hrow1 = hrow0 + d * HH;
    int ncol = tn * 16 + (lane & 15);
    int base = (lane & 16) ? 8 : 0;
    const float* fwr = fwT + (size_t)((l * HH + ncol) * 2) * HH;  // tap0; tap1 = +HH
    const float* gwr = gwT + (size_t)((l * HH + ncol) * 2) * HH;
    v8f cf = {}, cg = {};
    v16h a, bf, bg;
    // tap 0
    pack8(a, 0, hrow0 + base);  pack8(a, 8, hrow0 + base + 16);
    pack8(bf, 0, fwr + base);   pack8(bf, 8, fwr + base + 16);
    pack8(bg, 0, gwr + base);   pack8(bg, 8, gwr + base + 16);
    cf = wmma_f16(a, bf, cf);
    cg = wmma_f16(a, bg, cg);
    // tap 1
    pack8(a, 0, hrow1 + base);      pack8(a, 8, hrow1 + base + 16);
    pack8(bf, 0, fwr + HH + base);  pack8(bf, 8, fwr + HH + base + 16);
    pack8(bg, 0, gwr + HH + base);  pack8(bg, 8, gwr + HH + base + 16);
    cf = wmma_f16(a, bf, cf);
    cg = wmma_f16(a, bg, cg);

    int rbase = tm * 16 + ((lane & 16) ? 8 : 0);
    float fbias = fb[l * HH + ncol];
    float gbias = gb[l * HH + ncol];
#pragma unroll
    for (int v = 0; v < 8; v++) {
        int row = rbase + v;
        float fv = fast_tanh(cf[v] + fbias);
        float gv = fast_sigmoid(cg[v] + gbias);
        gau[(size_t)row * HH + ncol] = fv * gv;
    }
}

// ------------------ skip conv (last timestep only): skip += gau_last @ W^T + b
__global__ void k_skip_wmma(const float* __restrict__ gau, const float* __restrict__ sw,
                            const float* __restrict__ sb, float* skip, int Tp, int l) {
    const int nTilesN = SKC / 16;
    int tile = blockIdx.x * 4 + (threadIdx.x >> 5);
    int tm = tile / nTilesN, tn = tile % nTilesN;
    int lane = threadIdx.x & 31;
    int mrow = tm * 16 + (lane & 15);
    int ncol = tn * 16 + (lane & 15);
    int base = (lane & 16) ? 8 : 0;
    const float* arow = gau + (size_t)(mrow * Tp + (Tp - 1)) * HH;
    const float* brow = sw + (size_t)(l * SKC + ncol) * HH;
    v16h a, bm;
    pack8(a, 0, arow + base);  pack8(a, 8, arow + base + 16);
    pack8(bm, 0, brow + base); pack8(bm, 8, brow + base + 16);
    v8f c = {};
    c = wmma_f16(a, bm, c);
    int rbase = tm * 16 + ((lane & 16) ? 8 : 0);
    float bias = sb[l * SKC + ncol];
#pragma unroll
    for (int v = 0; v < 8; v++) {
        float* p = skip + (size_t)(rbase + v) * SKC + ncol;
        *p += c[v] + bias;
    }
}

// ------------------------------- message passing: agg[dst] += ew*ns[src]*gau[src]
__global__ void k_msg(const int* __restrict__ src, const int* __restrict__ dst,
                      const float* __restrict__ ew, const float* __restrict__ nsv,
                      const float* __restrict__ gau, float* agg, int Tp) {
    int idx = blockIdx.x * blockDim.x + threadIdx.x;
    int total = NE * BB * Tp * HH;
    if (idx >= total) return;
    int c = idx & 31;
    int t = (idx >> 5) % Tp;
    int r = idx / (Tp * HH);
    int e = r % NE;
    int b = r / NE;
    int s = src[e], d = dst[e];
    float coeff = ew[e] * nsv[s];
    float* pa = &agg[((b * NN + d) * Tp + t) * HH + c];
    __builtin_prefetch(pa, 1, 0);  // global_prefetch_b8: pull atomic target line
    atomicAdd(pa, coeff * gau[((b * NN + s) * Tp + t) * HH + c]);
}

// ---- gconv WMMA + bias + relu + residual, fused BN-stat accumulation
__global__ void k_gconv_wmma(const float* __restrict__ agg, const float* __restrict__ ndv,
                             const float* __restrict__ gcwT, const float* __restrict__ gcb,
                             const float* __restrict__ hold, float* hpre,
                             float* bnsum, float* bnsq, int T, int d, int l) {
    int Tp = T - d;
    int tile = blockIdx.x * 4 + (threadIdx.x >> 5);
    int tm = tile >> 1, tn = tile & 1;  // 2 N-tiles
    int lane = threadIdx.x & 31;
    int mrow = tm * 16 + (lane & 15);
    int node_m = (mrow / Tp) & 511;
    float nrm = ndv[node_m];
    int ncol = tn * 16 + (lane & 15);
    int base = (lane & 16) ? 8 : 0;
    const float* arow = agg + (size_t)mrow * HH;
    const float* brow = gcwT + (size_t)(l * HH + ncol) * HH;  // contiguous in k
    v16h a, bm;
    pack8_scale(a, 0, arow + base, nrm);
    pack8_scale(a, 8, arow + base + 16, nrm);
    pack8(bm, 0, brow + base);
    pack8(bm, 8, brow + base + 16);
    v8f c = {};
    c = wmma_f16(a, bm, c);
    int rbase = tm * 16 + ((lane & 16) ? 8 : 0);
    float bias = gcb[l * HH + ncol];
    float s1 = 0.f, s2 = 0.f;
#pragma unroll
    for (int v = 0; v < 8; v++) {
        int row = rbase + v;
        int t = row % Tp;
        int bn = row / Tp;  // b*NN + n
        float val = fmaxf(c[v] + bias, 0.f) + hold[(bn * T + t + d) * HH + ncol];
        hpre[(size_t)row * HH + ncol] = val;
        s1 += val;
        s2 += val * val;
    }
    atomicAdd(&bnsum[ncol], s1);
    atomicAdd(&bnsq[ncol], s2);
}

// ------------------------------- batchnorm normalize (training-mode batch stats)
__global__ void k_bn(float* h, const float* __restrict__ bnsum, const float* __restrict__ bnsq,
                     const float* __restrict__ gamma, const float* __restrict__ beta,
                     int Tp, int l) {
    int total = BB * NN * Tp * HH;
    int idx = blockIdx.x * blockDim.x + threadIdx.x;
    if (idx >= total) return;
    int c = idx & 31;
    float cnt = (float)(BB * NN * Tp);
    float mu = bnsum[c] / cnt;
    float var = bnsq[c] / cnt - mu * mu;
    float x = h[idx];
    h[idx] = (x - mu) * rsqrtf(var + 1e-5f) * gamma[l * HH + c] + beta[l * HH + c];
}

// ------------------------------- end1: relu(relu(skip) @ W1^T + b1); K=256
__global__ void k_end1_wmma(const float* __restrict__ skip, const float* __restrict__ w1,
                            const float* __restrict__ b1, float* out1) {
    const int nTilesN = ECC / 16;
    int tile = blockIdx.x * 4 + (threadIdx.x >> 5);
    int tm = tile / nTilesN, tn = tile % nTilesN;
    int lane = threadIdx.x & 31;
    int mrow = tm * 16 + (lane & 15);
    int ncol = tn * 16 + (lane & 15);
    int base = (lane & 16) ? 8 : 0;
    const float* arow = skip + (size_t)mrow * SKC;
    const float* brow = w1 + (size_t)ncol * SKC;
    v8f c = {};
#pragma unroll
    for (int k0 = 0; k0 < SKC; k0 += 32) {
        v16h a, bm;
        pack8_relu(a, 0, arow + k0 + base);
        pack8_relu(a, 8, arow + k0 + base + 16);
        pack8(bm, 0, brow + k0 + base);
        pack8(bm, 8, brow + k0 + base + 16);
        c = wmma_f16(a, bm, c);
    }
    int rbase = tm * 16 + ((lane & 16) ? 8 : 0);
    float bias = b1[ncol];
#pragma unroll
    for (int v = 0; v < 8; v++)
        out1[(size_t)(rbase + v) * ECC + ncol] = fmaxf(c[v] + bias, 0.f);
}

// ------------------------------- end2: out1 @ W2^T + b2 -> [B, 12, N]; K=512
__global__ void k_end2_wmma(const float* __restrict__ in1, const float* __restrict__ w2,
                            const float* __restrict__ b2, float* out) {
    int tm = blockIdx.x * 4 + (threadIdx.x >> 5);
    int lane = threadIdx.x & 31;
    int mrow = tm * 16 + (lane & 15);
    int ncol = lane & 15;  // 12 real outputs, cols 12-15 padded zero
    int base = (lane & 16) ? 8 : 0;
    const float* arow = in1 + (size_t)mrow * ECC;
    const float* brow = w2 + (size_t)ncol * ECC;
    v8f c = {};
#pragma unroll
    for (int k0 = 0; k0 < ECC; k0 += 32) {
        v16h a, bm = {};
        pack8(a, 0, arow + k0 + base);
        pack8(a, 8, arow + k0 + base + 16);
        if (ncol < OUTF) {
            pack8(bm, 0, brow + k0 + base);
            pack8(bm, 8, brow + k0 + base + 16);
        }
        c = wmma_f16(a, bm, c);
    }
    if (ncol < OUTF) {
        int rbase = tm * 16 + ((lane & 16) ? 8 : 0);
        float bias = b2[ncol];
#pragma unroll
        for (int v = 0; v < 8; v++) {
            int row = rbase + v;
            int b = row >> 9, n = row & 511;
            out[(b * OUTF + ncol) * NN + n] = c[v] + bias;
        }
    }
}

// ==========================================================================
extern "C" void kernel_launch(void* const* d_in, const int* in_sizes, int n_in,
                              void* d_out, int out_size, void* d_ws, size_t ws_size,
                              hipStream_t stream) {
    const float* feat     = (const float*)d_in[0];
    const int*   src      = (const int*)d_in[1];
    const int*   dst      = (const int*)d_in[2];
    const float* nv1      = (const float*)d_in[3];
    const float* nv2      = (const float*)d_in[4];
    const float* start_w  = (const float*)d_in[5];
    const float* start_b  = (const float*)d_in[6];
    const float* filter_w = (const float*)d_in[7];
    const float* filter_b = (const float*)d_in[8];
    const float* gate_w   = (const float*)d_in[9];
    const float* gate_b   = (const float*)d_in[10];
    const float* skip_w   = (const float*)d_in[11];
    const float* skip_b   = (const float*)d_in[12];
    const float* bn_gamma = (const float*)d_in[13];
    const float* bn_beta  = (const float*)d_in[14];
    const float* gconv_w  = (const float*)d_in[15];
    const float* gconv_b  = (const float*)d_in[16];
    const float* end1_w   = (const float*)d_in[17];
    const float* end1_b   = (const float*)d_in[18];
    const float* end2_w   = (const float*)d_in[19];
    const float* end2_b   = (const float*)d_in[20];

    float* ws = (float*)d_ws;
    size_t o = 0;
    unsigned char* adj = (unsigned char*)(ws + o); o += (NN * NN) / 4;  // 256 KB as bytes
    float* rowmax = ws + o; o += NN;
    float* rowsum = ws + o; o += NN;
    float* ew     = ws + o; o += NE;
    float* degout = ws + o; o += NN;
    float* degin  = ws + o; o += NN;
    float* nsv    = ws + o; o += NN;
    float* ndv    = ws + o; o += NN;
    float* bnsum  = ws + o; o += HH;
    float* bnsq   = ws + o; o += HH;
    float* fwT    = ws + o; o += NLAY * HH * 2 * HH;          // 16384
    float* gwT    = ws + o; o += NLAY * HH * 2 * HH;          // 16384
    float* gcwT   = ws + o; o += NLAY * HH * HH;              // 8192
    float* skip   = ws + o; o += (size_t)BB * NN * SKC;       // 4.19M
    float* hA     = ws + o; o += (size_t)BB * NN * 13 * HH;   // 6.82M
    float* hB     = ws + o; o += (size_t)BB * NN * 13 * HH;   // 6.82M
    float* gau    = ws + o; o += (size_t)BB * NN * 12 * HH;   // 6.29M
    float* agg    = ws + o; o += (size_t)BB * NN * 12 * HH;   // 6.29M
    float* out1   = gau;  // end1 output (8.39M) reuses gau+agg region (12.6M)

#define ZEROF(p, n) k_zero<<<((n) + 255) / 256, 256, 0, stream>>>((p), (n))

    ZEROF((float*)adj, (NN * NN) / 4);
    ZEROF(degout, NN);
    ZEROF(degin, NN);
    ZEROF(skip, BB * NN * SKC);

    k_prep_w<<<(NLAY * HH * 2 * HH + 255) / 256, 256, 0, stream>>>(filter_w, gate_w, gconv_w,
                                                                   fwT, gwT, gcwT);
    k_scatter<<<(NE + 255) / 256, 256, 0, stream>>>(src, dst, adj, degout, degin);
    k_rownorm<<<NN, 256, 0, stream>>>(adj, nv1, nv2, rowmax, rowsum);
    k_ew<<<(NE + 255) / 256, 256, 0, stream>>>(src, dst, nv1, nv2, rowmax, rowsum, ew);
    k_norms<<<(NN + 255) / 256, 256, 0, stream>>>(degout, degin, nsv, ndv);
    k_start<<<(BB * NN * 13 * HH + 255) / 256, 256, 0, stream>>>(feat, start_w, start_b, hA);

    const int dil[8] = {1, 2, 1, 2, 1, 2, 1, 2};
    float* hcur = hA;
    float* hnext = hB;
    int T = 13;
    for (int l = 0; l < NLAY; l++) {
        int d = dil[l];
        int Tp = T - d;
        int M = BB * NN * Tp;         // rows, divisible by 16
        int tilesGau = (M / 16) * 2;  // divisible by 4

        k_gau_wmma<<<tilesGau / 4, 128, 0, stream>>>(hcur, fwT, filter_b, gwT, gate_b,
                                                     gau, T, d, l);
        k_skip_wmma<<<(BB * NN / 16) * (SKC / 16) / 4, 128, 0, stream>>>(gau, skip_w, skip_b,
                                                                         skip, Tp, l);
        ZEROF(agg, M * HH);
        k_msg<<<(NE * BB * Tp * HH + 255) / 256, 256, 0, stream>>>(src, dst, ew, nsv,
                                                                   gau, agg, Tp);
        ZEROF(bnsum, HH);
        ZEROF(bnsq, HH);
        k_gconv_wmma<<<(M / 16) * 2 / 4, 128, 0, stream>>>(agg, ndv, gcwT, gconv_b,
                                                           hcur, hnext, bnsum, bnsq, T, d, l);
        k_bn<<<(M * HH + 255) / 256, 256, 0, stream>>>(hnext, bnsum, bnsq,
                                                       bn_gamma, bn_beta, Tp, l);
        float* tmp = hcur; hcur = hnext; hnext = tmp;
        T = Tp;
    }

    k_end1_wmma<<<(BB * NN / 16) * (ECC / 16) / 4, 128, 0, stream>>>(skip, end1_w, end1_b, out1);
    k_end2_wmma<<<(BB * NN / 16) / 4, 128, 0, stream>>>(out1, end2_w, end2_b, (float*)d_out);
#undef ZEROF
}
